// FidelityEncoder_2362232012938
// MI455X (gfx1250) — matrix operand order repaired and checked by reading, not compiled, not run on hardware
//
#include <hip/hip_runtime.h>
#include <math.h>

// ---------------------------------------------------------------------------
// FidelityEncoder (3-layer GAT + LN + global-attn pooling + MLP) for gfx1250.
// GEMMs use v_wmma_f32_16x16x32_f16 with split-f16 (hi/lo) operands for
// near-fp32 accuracy. Edge aggregation is CSR/dst-centric (no float atomics
// in the hot loop); everything node-sized lives in the 192MB L2.
// ---------------------------------------------------------------------------

typedef __attribute__((ext_vector_type(16))) _Float16 v16h;
typedef __attribute__((ext_vector_type(8)))  _Float16 v8h;
typedef __attribute__((ext_vector_type(8)))  float    v8f;
typedef __attribute__((ext_vector_type(4)))  float    v4f;

#define HDIM 128
#define WT_STRIDE 136   // padded LDS row stride (halves) -> conflict-light, 16B aligned
#define GEMM_SMEM (2 * 128 * WT_STRIDE * (int)sizeof(_Float16))  // 69632 B dynamic LDS

// ---------------- small utility kernels ----------------
__global__ void k_zero_i32(int* p, int n) {
  int i = blockIdx.x * blockDim.x + threadIdx.x;
  if (i < n) p[i] = 0;
}
__global__ void k_zero_f32(float* p, int n) {
  int i = blockIdx.x * blockDim.x + threadIdx.x;
  if (i < n) p[i] = 0.f;
}
__global__ void k_copy_i32(const int* __restrict__ a, int* __restrict__ b, int n) {
  int i = blockIdx.x * blockDim.x + threadIdx.x;
  if (i < n) b[i] = a[i];
}

__device__ __forceinline__ unsigned enc_f32(float f) {
  unsigned u = __float_as_uint(f);
  return (u & 0x80000000u) ? ~u : (u | 0x80000000u);   // monotonic float->uint
}
__device__ __forceinline__ float dec_f32(unsigned u) {
  unsigned v = (u & 0x80000000u) ? (u & 0x7fffffffu) : ~u;
  return __uint_as_float(v);
}

// ---------------- CSR build ----------------
__global__ void k_deg_count(const int* __restrict__ dst, int* __restrict__ deg, int E) {
  int e = blockIdx.x * blockDim.x + threadIdx.x;
  if (e < E) atomicAdd(&deg[dst[e]], 1);
}

// single-block exclusive scan over N elements (N ~ 50k -> trivial)
__global__ void k_scan(const int* __restrict__ deg, int* __restrict__ row_off, int n) {
  __shared__ int sh[1024];
  int carry = 0;
  for (int base = 0; base < n; base += 1024) {
    int i = base + (int)threadIdx.x;
    int v = (i < n) ? deg[i] : 0;
    sh[threadIdx.x] = v;
    __syncthreads();
    for (int off = 1; off < 1024; off <<= 1) {
      int t = (threadIdx.x >= (unsigned)off) ? sh[threadIdx.x - off] : 0;
      __syncthreads();
      sh[threadIdx.x] += t;
      __syncthreads();
    }
    if (i < n) row_off[i] = carry + sh[threadIdx.x] - v;  // exclusive
    int tot = sh[1023];
    __syncthreads();
    carry += tot;
  }
  if (threadIdx.x == 0) row_off[n] = carry;
}

__global__ void k_fill_csr(const int* __restrict__ src, const int* __restrict__ dst,
                           int* __restrict__ cursor, int* __restrict__ esrc, int E) {
  int e = blockIdx.x * blockDim.x + threadIdx.x;
  if (e < E) {
    int pos = atomicAdd(&cursor[dst[e]], 1);
    esrc[pos] = src[e];
  }
}

// ---------------- node embedding ----------------
__global__ void k_embed(const int* __restrict__ gt, const int* __restrict__ qi,
                        const float* __restrict__ gemb, const float* __restrict__ qemb,
                        float* __restrict__ h, int N) {
  int i = blockIdx.x * blockDim.x + threadIdx.x;
  if (i >= N * HDIM) return;
  int n = i >> 7, c = i & 127;
  h[i] = gemb[gt[n] * HDIM + c] + qemb[qi[n] * HDIM + c];
}

// ---------------- WMMA GEMM: feat[N,128] = h[N,128] @ W[128,128] ----------------
// Block = 8 waves; wave w owns rows [blk*128 + 16w, +16), all 8 column tiles.
// W staged transposed into LDS as split hi/lo f16; A fragments built from global.
__global__ void __launch_bounds__(256) k_gemm(const float* __restrict__ hin,
                                              const float* __restrict__ Wl,
                                              float* __restrict__ feat, int N) {
  extern __shared__ __align__(16) char smem[];
  _Float16* Wh = (_Float16*)smem;
  _Float16* Wo = Wh + 128 * WT_STRIDE;

  int tid = threadIdx.x;
  // stage W (row-major [k][n]) -> LDS transposed [n][k], split into hi/lo f16
  for (int i = tid; i < 128 * 128; i += 256) {
    int k = i >> 7, n = i & 127;
    float w = Wl[i];
    _Float16 hi = (_Float16)w;
    Wh[n * WT_STRIDE + k] = hi;
    Wo[n * WT_STRIDE + k] = (_Float16)(w - (float)hi);
  }
  __syncthreads();

  int lane = tid & 31, wv = tid >> 5;
  int mloc = lane & 15;
  int khalf = lane >> 4;                 // 0 or 1 (lane group)
  int rowbase = blockIdx.x * 128 + wv * 16;
  int row = rowbase + mloc;
  int rowc = (row < N) ? row : (N - 1);  // clamp reads: EXEC must stay all-1 for WMMA
  const float* hrow = hin + (size_t)rowc * HDIM;

  v8f acc[8];
#pragma unroll
  for (int t = 0; t < 8; t++)
#pragma unroll
    for (int j = 0; j < 8; j++) acc[t][j] = 0.f;

#pragma unroll
  for (int s = 0; s < 4; s++) {          // K steps of 32
    // A fragment: 16-bit A 16x32 layout: elems 0..7 -> K=abase..+7, 8..15 -> K=abase+16..+23
    int abase = s * 32 + khalf * 8;
    v4f a0 = *(const v4f*)(hrow + abase);
    v4f a1 = *(const v4f*)(hrow + abase + 4);
    v4f a2 = *(const v4f*)(hrow + abase + 16);
    v4f a3 = *(const v4f*)(hrow + abase + 20);
    float af[16];
#pragma unroll
    for (int j = 0; j < 4; j++) { af[j] = a0[j]; af[4 + j] = a1[j]; af[8 + j] = a2[j]; af[12 + j] = a3[j]; }
    v16h Ah, Al;
#pragma unroll
    for (int j = 0; j < 16; j++) {
      _Float16 hi = (_Float16)af[j];
      Ah[j] = hi;
      Al[j] = (_Float16)(af[j] - (float)hi);
    }
    int kb = s * 32 + khalf * 8;         // B 32x16 layout: elems 0..7 -> K=kb..+7, 8..15 -> K=kb+16..+23
#pragma unroll
    for (int ct = 0; ct < 8; ct++) {
      int ncol = ct * 16 + mloc;
      v8h bh0 = *(const v8h*)(Wh + ncol * WT_STRIDE + kb);
      v8h bh1 = *(const v8h*)(Wh + ncol * WT_STRIDE + kb + 16);
      v8h bl0 = *(const v8h*)(Wo + ncol * WT_STRIDE + kb);
      v8h bl1 = *(const v8h*)(Wo + ncol * WT_STRIDE + kb + 16);
      v16h Bh, Bl;
#pragma unroll
      for (int j = 0; j < 8; j++) { Bh[j] = bh0[j]; Bh[8 + j] = bh1[j]; Bl[j] = bl0[j]; Bl[8 + j] = bl1[j]; }
      // split-precision product: Ah*Bh + Al*Bh + Ah*Bl  (f32 accumulate)
      acc[ct] = __builtin_amdgcn_wmma_f32_16x16x32_f16(false, Ah, false, Bh, (short)0, acc[ct], false, false);
      acc[ct] = __builtin_amdgcn_wmma_f32_16x16x32_f16(false, Al, false, Bh, (short)0, acc[ct], false, false);
      acc[ct] = __builtin_amdgcn_wmma_f32_16x16x32_f16(false, Ah, false, Bl, (short)0, acc[ct], false, false);
    }
  }

  // C/D layout: VGPR r, lanes 0-15 -> M=r, lanes 16-31 -> M=r+8; N = lane&15
  int srow0 = rowbase + khalf * 8;
#pragma unroll
  for (int ct = 0; ct < 8; ct++) {
#pragma unroll
    for (int r = 0; r < 8; r++) {
      int rr = srow0 + r;
      if (rr < N) feat[(size_t)rr * HDIM + ct * 16 + mloc] = acc[ct][r];
    }
  }
}

// ---------------- per-head attention dots: el/er [N,4] ----------------
__global__ void __launch_bounds__(128) k_attn(const float* __restrict__ feat,
                                              const float* __restrict__ al,
                                              const float* __restrict__ ar,
                                              float* __restrict__ el, float* __restrict__ er, int N) {
  int n = blockIdx.x;
  int hd = threadIdx.x >> 5, o = threadIdx.x & 31;   // wave per head (wave32)
  float f = feat[(size_t)n * HDIM + hd * 32 + o];
  float sl = f * al[hd * 32 + o];
  float sr = f * ar[hd * 32 + o];
  for (int off = 16; off; off >>= 1) {
    sl += __shfl_xor(sl, off, 32);
    sr += __shfl_xor(sr, off, 32);
  }
  if (o == 0) { el[n * 4 + hd] = sl; er[n * 4 + hd] = sr; }
}

// ---------------- fused edge-softmax + aggregate + residual + relu ----------------
// One wave per dst node; lane owns channels [4*lane, 4*lane+4) (head = lane>>3).
__global__ void __launch_bounds__(256) k_aggr(const float* __restrict__ feat,
                                              const float* __restrict__ hin,
                                              const float* __restrict__ el,
                                              const float* __restrict__ er,
                                              const int* __restrict__ row_off,
                                              const int* __restrict__ esrc,
                                              const float* __restrict__ bias,
                                              float* __restrict__ hout, int N) {
  int wv = threadIdx.x >> 5, lane = threadIdx.x & 31;
  int n = blockIdx.x * 8 + wv;
  if (n >= N) return;
  int hd = lane >> 3;
  int cbase = lane * 4;
  int beg = row_off[n], end = row_off[n + 1];
  float ern = er[n * 4 + hd];

  float m = -3.4e38f;
  for (int i = beg; i < end; i++) {
    int s = esrc[i];
    float e = el[s * 4 + hd] + ern;
    e = (e > 0.f) ? e : 0.2f * e;               // leaky_relu(0.2)
    m = fmaxf(m, e);
  }
  float z = 0.f;
  float acc[4] = {0.f, 0.f, 0.f, 0.f};
  for (int i = beg; i < end; i++) {
    int s = esrc[i];
    float e = el[s * 4 + hd] + ern;
    e = (e > 0.f) ? e : 0.2f * e;
    float a = expf(e - m);
    z += a;
    v4f f = *(const v4f*)(feat + (size_t)s * HDIM + cbase);  // fully coalesced 512B row/wave
#pragma unroll
    for (int j = 0; j < 4; j++) acc[j] += a * f[j];
  }
  float inv = (end > beg) ? (1.f / z) : 0.f;
  size_t o = (size_t)n * HDIM + cbase;
#pragma unroll
  for (int j = 0; j < 4; j++) {
    float v = acc[j] * inv + hin[o + j] + bias[cbase + j];   // residual + bias
    hout[o + j] = (v > 0.f) ? v : 0.f;                       // relu
  }
}

// ---------------- LayerNorm (wave per node) ----------------
__global__ void __launch_bounds__(256) k_ln(const float* __restrict__ hin,
                                            const float* __restrict__ gamma,
                                            const float* __restrict__ beta,
                                            float* __restrict__ hout, int N) {
  int wv = threadIdx.x >> 5, lane = threadIdx.x & 31;
  int n = blockIdx.x * 8 + wv;
  if (n >= N) return;
  size_t o = (size_t)n * HDIM + lane * 4;
  v4f v = *(const v4f*)(hin + o);
  float s = v[0] + v[1] + v[2] + v[3];
  float s2 = v[0] * v[0] + v[1] * v[1] + v[2] * v[2] + v[3] * v[3];
  for (int off = 16; off; off >>= 1) {
    s += __shfl_xor(s, off, 32);
    s2 += __shfl_xor(s2, off, 32);
  }
  float mu = s * (1.f / 128.f);
  float var = s2 * (1.f / 128.f) - mu * mu;
  float r = rsqrtf(var + 1e-5f);
#pragma unroll
  for (int j = 0; j < 4; j++) {
    int c = lane * 4 + j;
    hout[o + j] = (v[j] - mu) * r * gamma[c] + beta[c];
  }
}

// ---------------- global attention pooling ----------------
__global__ void k_pool_init(unsigned* gmax, float* gz, int G) {
  int i = blockIdx.x * blockDim.x + threadIdx.x;
  if (i < G) { gmax[i] = enc_f32(-3.4e38f); gz[i] = 0.f; }
}

__global__ void __launch_bounds__(256) k_gate(const float* __restrict__ h,
                                              const float* __restrict__ gw,
                                              const float* __restrict__ gb,
                                              const int* __restrict__ gid,
                                              float* __restrict__ gate,
                                              unsigned* __restrict__ gmax, int N) {
  int wv = threadIdx.x >> 5, lane = threadIdx.x & 31;
  int n = blockIdx.x * 8 + wv;
  if (n >= N) return;
  size_t o = (size_t)n * HDIM + lane * 4;
  v4f v = *(const v4f*)(h + o);
  float p = 0.f;
#pragma unroll
  for (int j = 0; j < 4; j++) p += v[j] * gw[lane * 4 + j];
  for (int off = 16; off; off >>= 1) p += __shfl_xor(p, off, 32);
  if (lane == 0) {
    float g = p + gb[0];
    gate[n] = g;
    atomicMax(&gmax[gid[n]], enc_f32(g));
  }
}

__global__ void k_pool_a(const float* __restrict__ gate, const int* __restrict__ gid,
                         const unsigned* __restrict__ gmax, float* __restrict__ an,
                         float* __restrict__ gz, int N) {
  int n = blockIdx.x * blockDim.x + threadIdx.x;
  if (n >= N) return;
  int g = gid[n];
  float a = expf(gate[n] - dec_f32(gmax[g]));
  an[n] = a;
  atomicAdd(&gz[g], a);
}

__global__ void k_pool_sum(const float* __restrict__ h, const float* __restrict__ an,
                           const int* __restrict__ gid, float* __restrict__ zg, int N) {
  int i = blockIdx.x * blockDim.x + threadIdx.x;
  if (i >= N * HDIM) return;
  int n = i >> 7, c = i & 127;
  atomicAdd(&zg[gid[n] * HDIM + c], an[n] * h[i]);
}

// ---------------- projection MLP ----------------
__global__ void k_mlp1(const float* __restrict__ zg, const float* __restrict__ gz,
                       const float* __restrict__ w1, const float* __restrict__ b1,
                       float* __restrict__ z1, int G) {
  int i = blockIdx.x * blockDim.x + threadIdx.x;
  if (i >= G * HDIM) return;
  int g = i >> 7, j = i & 127;
  float inv = 1.f / gz[g];
  float s = b1[j];
  for (int c = 0; c < HDIM; c++) s += (zg[g * HDIM + c] * inv) * w1[c * HDIM + j];
  z1[i] = (s > 0.f) ? s : 0.f;
}

__global__ void k_mlp2(const float* __restrict__ z1, const float* __restrict__ w2,
                       const float* __restrict__ b2, float* __restrict__ out, int G) {
  int i = blockIdx.x * blockDim.x + threadIdx.x;
  if (i >= G * 64) return;
  int g = i >> 6, k = i & 63;
  float s = b2[k];
  for (int j = 0; j < HDIM; j++) s += z1[g * HDIM + j] * w2[j * 64 + k];
  out[i] = s;
}

// ---------------------------------------------------------------------------
extern "C" void kernel_launch(void* const* d_in, const int* in_sizes, int n_in,
                              void* d_out, int out_size, void* d_ws, size_t ws_size,
                              hipStream_t stream) {
  (void)n_in; (void)ws_size;
  const int N = in_sizes[0];
  const int E = in_sizes[2];
  const int G = out_size / 64;
  const int L = 3;

  const int*   gt    = (const int*)d_in[0];
  const int*   qi    = (const int*)d_in[1];
  const int*   src   = (const int*)d_in[2];
  const int*   dst   = (const int*)d_in[3];
  const int*   gid   = (const int*)d_in[4];
  const float* gemb  = (const float*)d_in[5];
  const float* qemb  = (const float*)d_in[6];
  const float* W     = (const float*)d_in[7];
  const float* attnl = (const float*)d_in[8];
  const float* attnr = (const float*)d_in[9];
  const float* bias  = (const float*)d_in[10];
  const float* gamma = (const float*)d_in[11];
  const float* beta  = (const float*)d_in[12];
  const float* gw    = (const float*)d_in[13];
  const float* gb    = (const float*)d_in[14];
  const float* w1    = (const float*)d_in[15];
  const float* b1    = (const float*)d_in[16];
  const float* w2    = (const float*)d_in[17];
  const float* b2    = (const float*)d_in[18];

  // --- workspace carving (bump allocator, 256B aligned) ---
  char* ws = (char*)d_ws;
  auto take = [&](size_t bytes) -> char* {
    char* p = ws;
    ws += (bytes + 255) & ~(size_t)255;
    return p;
  };
  float*    hA      = (float*)take((size_t)N * HDIM * 4);
  float*    hB      = (float*)take((size_t)N * HDIM * 4);
  float*    feat    = (float*)take((size_t)N * HDIM * 4);
  float*    el      = (float*)take((size_t)N * 4 * 4);
  float*    er      = (float*)take((size_t)N * 4 * 4);
  int*      deg     = (int*)take((size_t)N * 4);
  int*      row_off = (int*)take((size_t)(N + 1) * 4);
  int*      cursor  = (int*)take((size_t)N * 4);
  int*      esrc    = (int*)take((size_t)E * 4);
  float*    gate    = (float*)take((size_t)N * 4);
  float*    an      = (float*)take((size_t)N * 4);
  unsigned* gmax    = (unsigned*)take((size_t)G * 4);
  float*    gz      = (float*)take((size_t)G * 4);
  float*    zg      = (float*)take((size_t)G * HDIM * 4);
  float*    z1      = (float*)take((size_t)G * HDIM * 4);

  const int T = 256;
  // --- CSR build (graph is static across layers; built once per call) ---
  k_zero_i32<<<(N + T - 1) / T, T, 0, stream>>>(deg, N);
  k_deg_count<<<(E + T - 1) / T, T, 0, stream>>>(dst, deg, E);
  k_scan<<<1, 1024, 0, stream>>>(deg, row_off, N);
  k_copy_i32<<<(N + T - 1) / T, T, 0, stream>>>(row_off, cursor, N);
  k_fill_csr<<<(E + T - 1) / T, T, 0, stream>>>(src, dst, cursor, esrc, E);

  // --- embedding ---
  k_embed<<<((size_t)N * HDIM + T - 1) / T, T, 0, stream>>>(gt, qi, gemb, qemb, hA, N);

  // --- GAT layers ---
  float* hin = hA;
  float* hout = hB;
  for (int l = 0; l < L; l++) {
    k_gemm<<<(N + 127) / 128, 256, GEMM_SMEM, stream>>>(hin, W + (size_t)l * HDIM * HDIM, feat, N);
    k_attn<<<N, 128, 0, stream>>>(feat, attnl + l * HDIM, attnr + l * HDIM, el, er, N);
    k_aggr<<<(N + 7) / 8, 256, 0, stream>>>(feat, hin, el, er, row_off, esrc,
                                            bias + l * HDIM, hout, N);
    float* tmp = hin; hin = hout; hout = tmp;
  }

  // --- LayerNorm (hin -> hout) ---
  k_ln<<<(N + 7) / 8, 256, 0, stream>>>(hin, gamma, beta, hout, N);
  float* hn = hout;

  // --- global attention pooling ---
  k_pool_init<<<(G + 63) / 64, 64, 0, stream>>>(gmax, gz, G);
  k_zero_f32<<<(G * HDIM + T - 1) / T, T, 0, stream>>>(zg, G * HDIM);
  k_gate<<<(N + 7) / 8, 256, 0, stream>>>(hn, gw, gb, gid, gate, gmax, N);
  k_pool_a<<<(N + T - 1) / T, T, 0, stream>>>(gate, gid, gmax, an, gz, N);
  k_pool_sum<<<((size_t)N * HDIM + T - 1) / T, T, 0, stream>>>(hn, an, gid, zg, N);

  // --- projection MLP ---
  k_mlp1<<<(G * HDIM + T - 1) / T, T, 0, stream>>>(zg, gz, w1, b1, z1, G);
  k_mlp2<<<(G * 64 + T - 1) / T, T, 0, stream>>>(z1, w2, b2, (float*)d_out, G);
}
